// OrthogonalSAE_6871947674074
// MI455X (gfx1250) — matrix operand order repaired and checked by reading, not compiled, not run on hardware
//
#include <hip/hip_runtime.h>
#include <cstdint>
#include <cstddef>

// ---------------------------------------------------------------------------
// OrthogonalSAE forward for MI455X (gfx1250, wave32, WMMA).
// Pipeline:
//   1) inv column norms of W_enc, inv row norms of W_dec           (f32)
//   2) transpose+scale W_enc -> WencT bf16 [16384][2048]
//      transpose+scale W_dec -> WdecT bf16 [2048][16384]
//   3) acts f32 -> bf16
//   4) GEMM1 (WMMA bf16): pre = acts @ WencT^T + b_enc             (f32)
//   5) double layernorm per row (in place)
//   6) per-row radix-select: exact K-th largest -> thr
//   7) encode: bf16( h * sigmoid((h-thr)/T) * feature_mask )
//   8) GEMM2 (WMMA bf16): out = enc @ WdecT^T + b_dec              (f32)
// ---------------------------------------------------------------------------

#define M_ROWS   4096
#define D_IN     2048
#define D_SAE    16384
#define K_SEL    1638       // int(16384 * 0.1)
#define INV_TEMP 10.0f      // 1 / 0.1
#define W_EPS    1e-8f
#define LN_EPS   1e-5f

typedef __attribute__((ext_vector_type(16))) __bf16 v16bf;
typedef __attribute__((ext_vector_type(8)))  __bf16 v8bf;
typedef __attribute__((ext_vector_type(8)))  float  v8f;

// ---------------- helpers ----------------

__device__ inline v16bf cat8(v8bf lo, v8bf hi) {
    v16bf r;
#pragma unroll
    for (int i = 0; i < 8; ++i) { r[i] = lo[i]; r[i + 8] = hi[i]; }
    return r;
}

__device__ inline float block_reduce_sum(float v, float* sb, int tid) {
    sb[tid] = v; __syncthreads();
#pragma unroll
    for (int st = 128; st > 0; st >>= 1) {
        if (tid < st) sb[tid] += sb[tid + st];
        __syncthreads();
    }
    float r = sb[0]; __syncthreads();
    return r;
}

// order-preserving float<->uint mapping (larger float -> larger uint)
__device__ inline unsigned f2o(float f) {
    unsigned u = __float_as_uint(f);
    return (u & 0x80000000u) ? ~u : (u | 0x80000000u);
}
__device__ inline float o2f(unsigned u) {
    return (u & 0x80000000u) ? __uint_as_float(u ^ 0x80000000u)
                             : __uint_as_float(~u);
}

// ---------------- 1) norms ----------------

// inv_n[j] = 1/(||W_enc[:,j]|| + eps), coalesced over columns
__global__ __launch_bounds__(256) void col_inv_norm_kernel(
    const float* __restrict__ W, float* __restrict__ inv, int R, int C) {
    int j = blockIdx.x * 256 + threadIdx.x;
    if (j >= C) return;
    float s = 0.f;
    for (int i = 0; i < R; ++i) { float v = W[(size_t)i * C + j]; s += v * v; }
    inv[j] = 1.f / (sqrtf(s) + W_EPS);
}

// inv_m[i] = 1/(||W_dec[i,:]|| + eps), one wave32 per row
__global__ __launch_bounds__(256) void row_inv_norm_kernel(
    const float* __restrict__ W, float* __restrict__ inv, int R, int C) {
    int lane = threadIdx.x & 31;
    int wave = threadIdx.x >> 5;
    int row  = blockIdx.x * 8 + wave;
    if (row >= R) return;
    const float* p = W + (size_t)row * C;
    float s = 0.f;
    for (int j = lane; j < C; j += 32) { float v = p[j]; s += v * v; }
#pragma unroll
    for (int off = 16; off > 0; off >>= 1) s += __shfl_down(s, off, 32);
    if (lane == 0) inv[row] = 1.f / (sqrtf(s) + W_EPS);
}

// ---------------- 2) transpose + scale + bf16 ----------------
// src: R x C (f32), dst: C x R (bf16). scale_by_src_row=1 -> v *= inv[src row]
// scale_by_src_row=0 -> v *= inv[src col] (== dst row)
__global__ __launch_bounds__(256) void transpose_scale_bf16_kernel(
    const float* __restrict__ src, __bf16* __restrict__ dst,
    const float* __restrict__ inv, int R, int C, int scale_by_src_row) {
    __shared__ float tile[32][33];
    int tx = threadIdx.x & 31;
    int ty = threadIdx.x >> 5;            // 0..7
    int c0 = blockIdx.x * 32;
    int r0 = blockIdx.y * 32;
#pragma unroll
    for (int i = ty; i < 32; i += 8) {
        int r = r0 + i;
        float v = src[(size_t)r * C + c0 + tx];
        if (scale_by_src_row) v *= inv[r];
        tile[i][tx] = v;
    }
    __syncthreads();
#pragma unroll
    for (int i = ty; i < 32; i += 8) {
        int drow = c0 + i;                 // dst row == src column
        float sc = scale_by_src_row ? 1.f : inv[drow];
        dst[(size_t)drow * R + r0 + tx] = (__bf16)(tile[tx][i] * sc);
    }
}

// ---------------- 3) f32 -> bf16 ----------------

__global__ __launch_bounds__(256) void f32_to_bf16_kernel(
    const float* __restrict__ src, __bf16* __restrict__ dst, size_t n) {
    size_t i = (size_t)blockIdx.x * 256 + threadIdx.x;
    if (i < n) dst[i] = (__bf16)src[i];
}

// ---------------- 4/8) WMMA bf16 GEMM with bias ----------------
// C[M][N] = A[M][K] * BT[N][K]^T + bias[N]
// block = 256 threads = 8 waves (2x4), wave tile 64x64, block tile 128x256
__global__ __launch_bounds__(256) void wmma_gemm_bias_kernel(
    const __bf16* __restrict__ A, const __bf16* __restrict__ BT,
    const float* __restrict__ bias, float* __restrict__ C,
    int M, int N, int Kd) {
    const int lane = threadIdx.x & 31;
    const int wave = threadIdx.x >> 5;
    const int wm   = wave >> 2;               // 0..1
    const int wn   = wave & 3;                // 0..3
    const int m0   = blockIdx.y * 128 + wm * 64;
    const int n0   = blockIdx.x * 256 + wn * 64;
    const int l15  = lane & 15;
    const int sel  = lane >> 4;               // 0 or 1

    v8f acc[4][4];
#pragma unroll
    for (int i = 0; i < 4; ++i)
#pragma unroll
        for (int j = 0; j < 4; ++j)
#pragma unroll
            for (int r = 0; r < 8; ++r) acc[i][j][r] = 0.f;

    for (int k0 = 0; k0 < Kd; k0 += 32) {
        v16bf afrag[4], bfrag[4];
#pragma unroll
        for (int i = 0; i < 4; ++i) {
            // 16-bit A 16x32 layout: lane m holds K = 8*sel..+7 and 16+8*sel..+7
            const __bf16* ap = A + (size_t)(m0 + i * 16 + l15) * Kd + k0 + 8 * sel;
            v8bf lo = *(const v8bf*)ap;
            v8bf hi = *(const v8bf*)(ap + 16);
            afrag[i] = cat8(lo, hi);
        }
#pragma unroll
        for (int j = 0; j < 4; ++j) {
            // B 32x16 layout: lane n (half sel) holds K = 8*sel..+7 and 16+8*sel..+7
            const __bf16* bp = BT + (size_t)(n0 + j * 16 + l15) * Kd + k0 + 8 * sel;
            v8bf lo = *(const v8bf*)bp;
            v8bf hi = *(const v8bf*)(bp + 16);
            bfrag[j] = cat8(lo, hi);
        }
#pragma unroll
        for (int i = 0; i < 4; ++i)
#pragma unroll
            for (int j = 0; j < 4; ++j)
                acc[i][j] = __builtin_amdgcn_wmma_f32_16x16x32_bf16(
                    false, afrag[i], false, bfrag[j],
                    (short)0, acc[i][j], false, false);
    }

    // 32-bit C/D 16x16 layout: VGPR r -> M = 8*sel + r, N = l15
#pragma unroll
    for (int j = 0; j < 4; ++j) {
        int n = n0 + j * 16 + l15;
        float bv = bias[n];
#pragma unroll
        for (int i = 0; i < 4; ++i) {
#pragma unroll
            for (int r = 0; r < 8; ++r) {
                int m = m0 + i * 16 + sel * 8 + r;
                C[(size_t)m * N + n] = acc[i][j][r] + bv;
            }
        }
    }
}

// ---------------- 5) double layernorm (in place), one block per row ----------

__global__ __launch_bounds__(256) void double_ln_kernel(
    float* __restrict__ H, const float* __restrict__ lnw,
    const float* __restrict__ lnb, int N) {
    __shared__ float sb[256];
    const int tid = threadIdx.x;
    float* x = H + (size_t)blockIdx.x * N;

    float s = 0.f, ss = 0.f;
    for (int j = tid; j < N; j += 256) { float v = x[j]; s += v; ss += v * v; }
    float mu  = block_reduce_sum(s, sb, tid) / (float)N;
    float ex2 = block_reduce_sum(ss, sb, tid) / (float)N;
    float rs0 = rsqrtf(ex2 - mu * mu + LN_EPS);

    float s1 = 0.f, ss1 = 0.f;
    for (int j = tid; j < N; j += 256) {
        float v = (x[j] - mu) * rs0 * lnw[j] + lnb[j];
        s1 += v; ss1 += v * v;
    }
    float mu1  = block_reduce_sum(s1, sb, tid) / (float)N;
    float ex21 = block_reduce_sum(ss1, sb, tid) / (float)N;
    float rs1  = rsqrtf(ex21 - mu1 * mu1 + LN_EPS);

    for (int j = tid; j < N; j += 256) {
        float v = (x[j] - mu) * rs0 * lnw[j] + lnb[j];
        x[j] = (v - mu1) * rs1;
    }
}

// ---------------- 6) exact K-th largest per row via radix select -------------

__global__ __launch_bounds__(256) void topk_threshold_kernel(
    const float* __restrict__ H, float* __restrict__ thr, int N, int Ksel) {
    __shared__ unsigned hist[256];
    __shared__ unsigned s_pref, s_pmask;
    __shared__ int s_krem;
    const int tid = threadIdx.x;
    const float* x = H + (size_t)blockIdx.x * N;

    if (tid == 0) { s_pref = 0u; s_pmask = 0u; s_krem = Ksel; }
    __syncthreads();

    for (int pass = 0; pass < 4; ++pass) {
        int shift = 24 - 8 * pass;
        hist[tid] = 0u;
        __syncthreads();
        unsigned pref = s_pref, pmask = s_pmask;
        for (int j = tid; j < N; j += 256) {
            unsigned u = f2o(x[j]);
            if ((u & pmask) == pref) atomicAdd(&hist[(u >> shift) & 0xFFu], 1u);
        }
        __syncthreads();
        if (tid == 0) {
            int krem = s_krem;
            unsigned cum = 0u;
            int b = 255;
            for (; b > 0; --b) {
                unsigned c = hist[b];
                if (cum + c >= (unsigned)krem) break;
                cum += c;
            }
            s_krem  = krem - (int)cum;
            s_pref  = pref | ((unsigned)b << shift);
            s_pmask = pmask | (0xFFu << shift);
        }
        __syncthreads();
    }
    if (tid == 0) thr[blockIdx.x] = o2f(s_pref);
}

// ---------------- 7) encode: soft-mask + feature mask -> bf16 ----------------

__global__ __launch_bounds__(256) void encode_kernel(
    const float* __restrict__ H, const float* __restrict__ thr,
    const float* __restrict__ fmask, __bf16* __restrict__ enc) {
    size_t idx = (size_t)blockIdx.x * 256 + threadIdx.x;
    float h = H[idx];
    float t = thr[idx >> 14];                 // D_SAE = 2^14
    float sg = 1.f / (1.f + __expf(-(h - t) * INV_TEMP));
    enc[idx] = (__bf16)(h * sg * fmask[idx & (D_SAE - 1)]);
}

// ---------------- launch ----------------

extern "C" void kernel_launch(void* const* d_in, const int* in_sizes, int n_in,
                              void* d_out, int out_size, void* d_ws, size_t ws_size,
                              hipStream_t stream) {
    (void)in_sizes; (void)n_in; (void)out_size; (void)ws_size;

    const float* acts  = (const float*)d_in[0];
    const float* W_enc = (const float*)d_in[1];
    const float* W_dec = (const float*)d_in[2];
    const float* b_enc = (const float*)d_in[3];
    const float* b_dec = (const float*)d_in[4];
    const float* ln_w  = (const float*)d_in[5];
    const float* ln_b  = (const float*)d_in[6];
    const float* fmask = (const float*)d_in[7];
    float* out = (float*)d_out;

    char* ws = (char*)d_ws;
    constexpr size_t O_INVN  = 0;
    constexpr size_t O_INVM  = O_INVN  + (size_t)D_SAE * 4;
    constexpr size_t O_ACTS  = O_INVM  + (size_t)D_SAE * 4;
    constexpr size_t O_WENCT = O_ACTS  + (size_t)M_ROWS * D_IN  * 2;
    constexpr size_t O_WDECT = O_WENCT + (size_t)D_SAE  * D_IN  * 2;
    constexpr size_t O_PRE   = O_WDECT + (size_t)D_IN   * D_SAE * 2;
    constexpr size_t O_THR   = O_PRE   + (size_t)M_ROWS * D_SAE * 4;
    constexpr size_t O_ENC   = O_THR   + 65536;

    float*  inv_n = (float*)(ws + O_INVN);
    float*  inv_m = (float*)(ws + O_INVM);
    __bf16* a_bf  = (__bf16*)(ws + O_ACTS);
    __bf16* wencT = (__bf16*)(ws + O_WENCT);
    __bf16* wdecT = (__bf16*)(ws + O_WDECT);
    float*  pre   = (float*)(ws + O_PRE);
    float*  thr   = (float*)(ws + O_THR);
    __bf16* enc   = (__bf16*)(ws + O_ENC);

    // 1) norms
    col_inv_norm_kernel<<<D_SAE / 256, 256, 0, stream>>>(W_enc, inv_n, D_IN, D_SAE);
    row_inv_norm_kernel<<<D_SAE / 8,   256, 0, stream>>>(W_dec, inv_m, D_SAE, D_IN);

    // 2) transposed, scaled bf16 weights
    transpose_scale_bf16_kernel<<<dim3(D_SAE / 32, D_IN / 32), 256, 0, stream>>>(
        W_enc, wencT, inv_n, D_IN, D_SAE, /*scale_by_src_row=*/0);
    transpose_scale_bf16_kernel<<<dim3(D_IN / 32, D_SAE / 32), 256, 0, stream>>>(
        W_dec, wdecT, inv_m, D_SAE, D_IN, /*scale_by_src_row=*/1);

    // 3) acts -> bf16
    f32_to_bf16_kernel<<<(M_ROWS * D_IN) / 256, 256, 0, stream>>>(
        acts, a_bf, (size_t)M_ROWS * D_IN);

    // 4) GEMM1: pre = acts @ W_enc_n + b_enc
    wmma_gemm_bias_kernel<<<dim3(D_SAE / 256, M_ROWS / 128), 256, 0, stream>>>(
        a_bf, wencT, b_enc, pre, M_ROWS, D_SAE, D_IN);

    // 5) double layernorm (in place)
    double_ln_kernel<<<M_ROWS, 256, 0, stream>>>(pre, ln_w, ln_b, D_SAE);

    // 6) exact top-K threshold
    topk_threshold_kernel<<<M_ROWS, 256, 0, stream>>>(pre, thr, D_SAE, K_SEL);

    // 7) encode -> bf16
    encode_kernel<<<(size_t)M_ROWS * D_SAE / 256, 256, 0, stream>>>(
        pre, thr, fmask, enc);

    // 8) GEMM2: out = enc @ W_dec_n + b_dec
    wmma_gemm_bias_kernel<<<dim3(D_IN / 256, M_ROWS / 128), 256, 0, stream>>>(
        enc, wdecT, b_dec, out, M_ROWS, D_IN, D_SAE);
}
